// HeuristicModel9_62869731279355
// MI455X (gfx1250) — matrix operand order repaired and checked by reading, not compiled, not run on hardware
//
#include <hip/hip_runtime.h>

typedef __attribute__((ext_vector_type(16))) _Float16 v16h;
typedef __attribute__((ext_vector_type(8)))  float    v8f;

// window score table and dscore/5 = low + 1000*high split tables (all f16-exact)
__constant__ float c_TBL[6] = {0.f, 5.f, 50.f, 500.f, 5000.f, 100000.f};
__constant__ int   c_DL[6]  = {1, 9, 90, 900, 0, 0};     // oc==0: (T[pc+1]-T[pc])/5 low part
__constant__ int   c_DH[6]  = {0, 0, 0, 0, 19, 0};       // oc==0: high part (x1000)
__constant__ int   c_ML[6]  = {0, 1, 10, 100, 1000, 0};  // pc==0&&oc>0: T[oc]/5 low part
__constant__ int   c_MH[6]  = {0, 0, 0, 0, 0, 20};       // pc==0&&oc>0: high part (x1000)

// 16-bit A-matrix 16x32 element->K map (ISA 7.12.2): lanes 0-15 K=0..7,16..23 ; lanes 16-31 +8
__device__ __forceinline__ int kmapA(int e, int h) {
  int r = e >> 1, p = e & 1;
  return 2 * r + ((r >= 4) ? 8 : 0) + p + 8 * h;
}

// 5-in-a-row windows on 9x9, same enumeration order as the reference
__device__ __forceinline__ void window_cells(int w, int* cells) {
  int dr, dc, r, c;
  if (w < 45)       { dr = 0; dc = 1;  r = w / 5;  c = w % 5; }
  else if (w < 90)  { int t = w - 45;  dr = 1; dc = 0;  r = t / 9; c = t % 9; }
  else if (w < 115) { int t = w - 90;  dr = 1; dc = 1;  r = t / 5; c = t % 5; }
  else              { int t = w - 115; dr = 1; dc = -1; r = t / 5; c = t % 5 + 4; }
#pragma unroll
  for (int k = 0; k < 5; ++k) cells[k] = (r + k * dr) * 9 + (c + k * dc);
}

union PackLog {
  unsigned int pack[64 * 160];  // phase1 -> phase2: packed (low|high<<16|win<<24) per (board,window)
  float        logits[64 * 96]; // phase2 -> softmax (aliased after pack is consumed)
};

__global__ __launch_bounds__(128) void heuristic9_kernel(
    const float* __restrict__ boards, const int* __restrict__ player,
    float* __restrict__ out, int Btot)
{
  __shared__ unsigned char s_board[64 * 96];   // cell values 0/1/2, pad=255
  __shared__ unsigned char s_player[64];
  __shared__ unsigned int  s_wmask[160][4];    // 96-bit membership mask per window (3 words used)
  __shared__ float         s_ev[64];
  __shared__ PackLog       s_u;

  const int tid  = threadIdx.x;
  const int wv   = tid >> 5;        // wave id: 16 boards per wave
  const int lane = tid & 31;
  const int ln   = lane & 15;       // A-row / B-col / D-col index
  const int h    = lane >> 4;       // half-wave select
  const int b0   = blockIdx.x * 64; // first board of this workgroup

  // ---------------- staging ----------------
  for (int i = tid; i < 64 * 96; i += 128) {
    int b = i / 96, cell = i - b * 96;
    unsigned char v = 255;
    if (cell < 81) v = (unsigned char)boards[(size_t)(b0 + b) * 81 + cell];
    s_board[i] = v;
  }
  if (tid < 64) {
    s_player[tid] = (unsigned char)player[b0 + tid];
    s_ev[tid] = 0.f;
  }
  for (int i = tid; i < 160; i += 128) {
    unsigned int m0 = 0, m1 = 0, m2 = 0;
    if (i < 140) {
      int cells[5]; window_cells(i, cells);
#pragma unroll
      for (int k = 0; k < 5; ++k) {
        int cc = cells[k];
        if (cc < 32)      m0 |= 1u << cc;
        else if (cc < 64) m1 |= 1u << (cc - 32);
        else              m2 |= 1u << (cc - 64);
      }
    }
    s_wmask[i][0] = m0; s_wmask[i][1] = m1; s_wmask[i][2] = m2;
  }
  for (int i = tid; i < 64 * 160; i += 128) s_u.pack[i] = 0u;
  __syncthreads();

  // ---------------- phase 1: PC/OC = Ind @ MEMBER via WMMA f16 ----------------
  const int bl = wv * 16 + ln;              // board owned as A-row
  const int pl = s_player[bl];
  const int op = 3 - pl;

  v16h aP[3], aO[3];                        // 16x96 indicator fragments (K = cells)
#pragma unroll
  for (int kc = 0; kc < 3; ++kc) {
#pragma unroll
    for (int e = 0; e < 16; ++e) {
      int cell = kmapA(e, h) + 32 * kc;     // < 96
      int v = s_board[bl * 96 + cell];
      aP[kc][e] = (_Float16)((v == pl) ? 1.f : 0.f);
      aO[kc][e] = (_Float16)((v == op) ? 1.f : 0.f);
    }
  }

  float evp[8];
#pragma unroll
  for (int r = 0; r < 8; ++r) evp[r] = 0.f;

#pragma unroll
  for (int nt = 0; nt < 9; ++nt) {          // 144 window slots (140 real)
    v8f cp = {}, co = {};
    const int wn = ln + 16 * nt;
#pragma unroll
    for (int kc = 0; kc < 3; ++kc) {
      v16h bfr;                             // MEMBER tile: K = cell, N = window
#pragma unroll
      for (int e = 0; e < 16; ++e) {
        int cell = kmapA(e, h) + 32 * kc;
        bfr[e] = (_Float16)(float)((s_wmask[wn][cell >> 5] >> (cell & 31)) & 1u);
      }
      cp = __builtin_amdgcn_wmma_f32_16x16x32_f16(false, aP[kc], false, bfr, (short)0, cp, false, false);
      co = __builtin_amdgcn_wmma_f32_16x16x32_f16(false, aO[kc], false, bfr, (short)0, co, false, false);
    }
#pragma unroll
    for (int r = 0; r < 8; ++r) {
      if (wn < 140) {
        int pc = (int)(cp[r] + 0.5f);
        int oc = (int)(co[r] + 0.5f);
        float plus  = (oc == 0 && pc > 0) ? c_TBL[pc] : 0.f;
        float minus = (pc == 0 && oc > 0) ? c_TBL[oc] : 0.f;
        evp[r] += plus - minus;
        int low = 0, high = 0;
        if (oc == 0)      { low = c_DL[pc]; high = c_DH[pc]; }
        else if (pc == 0) { low = c_ML[oc]; high = c_MH[oc]; }
        int winf = (pc == 4) ? 1 : 0;
        unsigned int pk = (unsigned)low | ((unsigned)high << 16) | ((unsigned)winf << 24);
        s_u.pack[(wv * 16 + r + 8 * h) * 160 + wn] = pk;
      }
    }
  }
#pragma unroll
  for (int r = 0; r < 8; ++r)
    atomicAdd(&s_ev[wv * 16 + r + 8 * h], evp[r]);   // ds_add_f32
  __syncthreads();

  // ---------------- phase 2: score/win = DSCORE @ MEMBER^T via WMMA f16 ----------------
  v16h aL[5], aH[5], aW[5];                 // 16x160 dscore fragments (K = windows)
#pragma unroll
  for (int kc = 0; kc < 5; ++kc) {
#pragma unroll
    for (int e = 0; e < 16; ++e) {
      int wn = kmapA(e, h) + 32 * kc;       // < 160
      unsigned int pk = s_u.pack[bl * 160 + wn];
      aL[kc][e] = (_Float16)(float)(pk & 0xFFFFu);
      aH[kc][e] = (_Float16)(float)((pk >> 16) & 0xFFu);
      aW[kc][e] = (_Float16)(float)(pk >> 24);
    }
  }

  float lg[6][8];
#pragma unroll
  for (int nt = 0; nt < 6; ++nt) {          // 96 cell slots (81 real)
    v8f cL = {}, cH = {}, cW = {};
    const int cell = ln + 16 * nt;
#pragma unroll
    for (int kc = 0; kc < 5; ++kc) {
      v16h bfr;                             // MEMBER^T tile: K = window, N = cell
#pragma unroll
      for (int e = 0; e < 16; ++e) {
        int wn = kmapA(e, h) + 32 * kc;
        bfr[e] = (_Float16)(float)((s_wmask[wn][cell >> 5] >> (cell & 31)) & 1u);
      }
      cL = __builtin_amdgcn_wmma_f32_16x16x32_f16(false, aL[kc], false, bfr, (short)0, cL, false, false);
      cH = __builtin_amdgcn_wmma_f32_16x16x32_f16(false, aH[kc], false, bfr, (short)0, cH, false, false);
      cW = __builtin_amdgcn_wmma_f32_16x16x32_f16(false, aW[kc], false, bfr, (short)0, cW, false, false);
    }
#pragma unroll
    for (int r = 0; r < 8; ++r) {
      int bd = wv * 16 + r + 8 * h;
      float score = s_ev[bd] + 5.f * (cL[r] + 1000.f * cH[r]);
      bool winb  = cW[r] > 0.5f;
      bool empty = (cell < 81) && (s_board[bd * 96 + cell] == 0);
      lg[nt][r] = empty ? (winb ? 100000.f : score) : -1000000.f;
    }
  }
  __syncthreads();                           // everyone done reading s_u.pack
#pragma unroll
  for (int nt = 0; nt < 6; ++nt)
#pragma unroll
    for (int r = 0; r < 8; ++r)
      s_u.logits[(wv * 16 + r + 8 * h) * 96 + (ln + 16 * nt)] = lg[nt][r];
  __syncthreads();

  // ---------------- softmax over 81 cells + value head ----------------
  if (tid < 64) {
    const int bg = b0 + tid;
    const float* lgp = &s_u.logits[tid * 96];
    float mx = -1e30f;
    for (int c2 = 0; c2 < 81; ++c2) mx = fmaxf(mx, lgp[c2]);
    float sum = 0.f;
    for (int c2 = 0; c2 < 81; ++c2) sum += __expf(0.03f * (lgp[c2] - mx));
    float inv = 1.f / sum;
    for (int c2 = 0; c2 < 81; ++c2)
      out[(size_t)bg * 81 + c2] = __expf(0.03f * (lgp[c2] - mx)) * inv;
    float val = tanhf(s_ev[tid] * (1.f / 3000.f));
    val = fminf(0.95f, fmaxf(-0.95f, val));
    out[(size_t)Btot * 81 + bg] = val;
  }
}

extern "C" void kernel_launch(void* const* d_in, const int* in_sizes, int n_in,
                              void* d_out, int out_size, void* d_ws, size_t ws_size,
                              hipStream_t stream) {
  const float* boards = (const float*)d_in[0];
  const int*   player = (const int*)d_in[1];
  float* out = (float*)d_out;
  int Btot = in_sizes[0] / 81;               // 2048
  int grid = Btot / 64;                      // 64 boards per 128-thread workgroup
  heuristic9_kernel<<<grid, 128, 0, stream>>>(boards, player, out, Btot);
}
